// DoubleAttentionLayer_57406532878344
// MI455X (gfx1250) — matrix operand order repaired and checked by reading, not compiled, not run on hardware
//
#include <hip/hip_runtime.h>
#include <hip/hip_bf16.h>
#include <math.h>

typedef __attribute__((ext_vector_type(16))) __bf16 v16bf;
typedef __attribute__((ext_vector_type(8)))  __bf16 v8bf;
typedef __attribute__((ext_vector_type(4)))  __bf16 v4bf;
typedef __attribute__((ext_vector_type(8)))  float  v8f;

#define S_LEN 1024
#define EMBD  512
#define NHEAD 8
#define HDIM  64
#define BATCH 2
#define MROWS (BATCH * S_LEN)   // 2048

static __device__ __forceinline__ __bf16 f2bf(float x) { return (__bf16)x; }

// ---------------------------------------------------------------------------
// Prep: f32 -> bf16 elementwise (n multiple of 1024; one float4 per thread)
// ---------------------------------------------------------------------------
__global__ __launch_bounds__(256)
void cvt_bf16_kernel(const float* __restrict__ src, __bf16* __restrict__ dst)
{
    const int t = blockIdx.x * 256 + threadIdx.x;
    float4 v = ((const float4*)src)[t];
    v4bf b = { f2bf(v.x), f2bf(v.y), f2bf(v.z), f2bf(v.w) };
    ((v4bf*)dst)[t] = b;
}

// ---------------------------------------------------------------------------
// Prep: transpose W(K x N) f32 -> Wt(N x K) bf16, 32x32 tiles via LDS
// ---------------------------------------------------------------------------
__global__ __launch_bounds__(256)
void transpose_bf16_kernel(const float* __restrict__ W, __bf16* __restrict__ Wt,
                           int K, int N)
{
    __shared__ float tile[32][33];
    const int tid = threadIdx.x;
    const int n0 = blockIdx.x * 32;
    const int k0 = blockIdx.y * 32;

    const int rk = tid >> 3;
    const int cn = (tid & 7) * 4;
    float4 v = *(const float4*)&W[(size_t)(k0 + rk) * N + n0 + cn];
    tile[rk][cn + 0] = v.x;
    tile[rk][cn + 1] = v.y;
    tile[rk][cn + 2] = v.z;
    tile[rk][cn + 3] = v.w;
    __syncthreads();

    const int rn = tid >> 3;
    const int ck = (tid & 7) * 4;
    v4bf b = { f2bf(tile[ck + 0][rn]), f2bf(tile[ck + 1][rn]),
               f2bf(tile[ck + 2][rn]), f2bf(tile[ck + 3][rn]) };
    *(v4bf*)&Wt[(size_t)(n0 + rn) * K + k0 + ck] = b;
}

// ---------------------------------------------------------------------------
// Projection GEMM: C(2048 x N) = A(2048 x Ka) @ W(Ka x N).
// A is bf16 row-major, Wt is bf16 transposed (N x Ka) -> all fragments are
// direct global vector loads; no LDS, no barriers. One wave = 16x64 tile.
// Epilogue applies bias(es) + tanh, scatters to head-major bf16 tensors.
// mode 0/1: two outputs tanh(x + wb + hb1/hb2) -> out1/out2 (B*NH, S, 64)
// mode 2  : kv split -> out1 kh (B*NH,S,64) for n<512; out2 vt (B*NH,64,S)
// mode 3  : single out1 = tanh(x + wb)  (kp, (B*NH,S,64))
// ---------------------------------------------------------------------------
__global__ __launch_bounds__(32)
void proj_gemm_kernel(const __bf16* __restrict__ A, const __bf16* __restrict__ Wt,
                      const float* __restrict__ wb,
                      const float* __restrict__ hb1, const float* __restrict__ hb2,
                      __bf16* __restrict__ out1, __bf16* __restrict__ out2,
                      int Ka, int mode)
{
    const int lane = threadIdx.x;
    const int lo = lane & 15;
    const int hi = lane >> 4;
    const int n0 = blockIdx.x * 64;
    const int m0 = blockIdx.y * 16;

    v8f zf = {};
    v8f acc[4] = { zf, zf, zf, zf };

    const __bf16* arow = A + (size_t)(m0 + lo) * Ka;

    for (int kk = 0; kk < Ka; kk += 32) {
        // A fragment: m = lo, k = 16hi + s (contiguous 16)
        v8bf a0 = *(const v8bf*)(arow + kk + hi * 16);
        v8bf a1 = *(const v8bf*)(arow + kk + hi * 16 + 8);
        v16bf af;
        #pragma unroll
        for (int i = 0; i < 8; ++i) { af[i] = a0[i]; af[8 + i] = a1[i]; }
        // B fragments: n = lo (+16g), k = 16hi + s (contiguous 16)
        #pragma unroll
        for (int g = 0; g < 4; ++g) {
            v16bf bf = *(const v16bf*)(Wt + (size_t)(n0 + g * 16 + lo) * Ka + kk + hi * 16);
            acc[g] = __builtin_amdgcn_wmma_f32_16x16x32_bf16(false, af, false, bf,
                                                             (short)0, acc[g], false, false);
        }
    }

    // Epilogue: D element r of tile g -> (m = m0 + r + 8*hi, n = n0 + 16g + lo)
    #pragma unroll
    for (int g = 0; g < 4; ++g) {
        const int n = n0 + g * 16 + lo;
        const float base = wb[n];
        const float b1 = (mode <= 1) ? hb1[n] : 0.0f;
        const float b2 = (mode <= 1) ? hb2[n] : 0.0f;
        #pragma unroll
        for (int r = 0; r < 8; ++r) {
            const int m = m0 + r + hi * 8;
            const int bb = m >> 10;            // batch
            const int ss = m & (S_LEN - 1);    // seq pos
            const float v = acc[g][r] + base;
            if (mode <= 1) {
                const int h = n >> 6, d = n & 63;
                const size_t idx = (((size_t)(bb * NHEAD + h)) * S_LEN + ss) * HDIM + d;
                out1[idx] = f2bf(tanhf(v + b1));
                out2[idx] = f2bf(tanhf(v + b2));
            } else if (mode == 3 || n < EMBD) {
                const int h = n >> 6, d = n & 63;
                const size_t idx = (((size_t)(bb * NHEAD + h)) * S_LEN + ss) * HDIM + d;
                out1[idx] = f2bf(tanhf(v));
            } else {
                const int nn = n - EMBD;
                const int h = nn >> 6, d = nn & 63;
                // V stored d-major: (B*NH, HD, S)
                const size_t idx = (((size_t)(bb * NHEAD + h)) * HDIM + d) * S_LEN + ss;
                out2[idx] = f2bf(tanhf(v));
            }
        }
    }
}

// ---------------------------------------------------------------------------
// Flash attention per (b, head, 16-query tile). Keys processed in blocks of 32.
// rel_shift folded in: pos-query row = j + (b==0 ? 1 : 0), zero when OOB.
// Causal mask applied analytically (key > query -> -1e30).
// qc/qp/kh/kp: (B*NH, S, 64) bf16;  vt: (B*NH, 64, S) bf16 (d-major).
// ---------------------------------------------------------------------------
__global__ __launch_bounds__(32)
void attn_kernel(const __bf16* __restrict__ qc, const __bf16* __restrict__ qp,
                 const __bf16* __restrict__ kh, const __bf16* __restrict__ kp,
                 const __bf16* __restrict__ vt, float* __restrict__ outp)
{
    __shared__ __align__(32) __bf16 p_lds[16 * 32];   // exp'd scores, [m][k]

    const int lane = threadIdx.x;
    const int lo = lane & 15;
    const int hi = lane >> 4;
    const int jt = blockIdx.x;
    const int h  = blockIdx.y;
    const int b  = blockIdx.z;
    const size_t headBase = (size_t)(b * NHEAD + h) * S_LEN * HDIM;

    // --- Load Q fragments (fixed for the whole key loop) ---
    v16bf qcf[2], qpf[2];
    {
        const int jq = jt * 16 + lo;
        const __bf16* qrow = qc + headBase + (size_t)jq * HDIM;
        #pragma unroll
        for (int c = 0; c < 2; ++c) {
            v8bf l0 = *(const v8bf*)(qrow + c * 32 + hi * 16);
            v8bf l1 = *(const v8bf*)(qrow + c * 32 + hi * 16 + 8);
            #pragma unroll
            for (int i = 0; i < 8; ++i) { qcf[c][i] = l0[i]; qcf[c][8 + i] = l1[i]; }
        }
        const int j2 = jq + ((b == 0) ? 1 : 0);   // rel_shift row offset
        if (j2 < S_LEN) {
            const __bf16* prow = qp + headBase + (size_t)j2 * HDIM;
            #pragma unroll
            for (int c = 0; c < 2; ++c) {
                v8bf l0 = *(const v8bf*)(prow + c * 32 + hi * 16);
                v8bf l1 = *(const v8bf*)(prow + c * 32 + hi * 16 + 8);
                #pragma unroll
                for (int i = 0; i < 8; ++i) { qpf[c][i] = l0[i]; qpf[c][8 + i] = l1[i]; }
            }
        } else {
            #pragma unroll
            for (int c = 0; c < 2; ++c)
                #pragma unroll
                for (int i = 0; i < 16; ++i) qpf[c][i] = f2bf(0.0f);
        }
    }

    v8f zf = {};
    v8f oacc[4] = { zf, zf, zf, zf };
    float mrun[8], lrun[8];
    #pragma unroll
    for (int r = 0; r < 8; ++r) { mrun[r] = -1e30f; lrun[r] = 0.0f; }

    const int nb = (jt * 16 + 15) / 32 + 1;   // causal: key blocks needed

    for (int ib = 0; ib < nb; ++ib) {
        const int kb = ib * 32;

        // --- Score tile S(16x32) = Qc.Kh^T + Qp'.Kp^T ---
        v8f sacc[2] = { zf, zf };
        #pragma unroll
        for (int t = 0; t < 2; ++t) {
            const __bf16* krow = kh + headBase + (size_t)(kb + t * 16 + lo) * HDIM;
            const __bf16* prow = kp + headBase + (size_t)(kb + t * 16 + lo) * HDIM;
            #pragma unroll
            for (int c = 0; c < 2; ++c) {
                v16bf kf = *(const v16bf*)(krow + c * 32 + hi * 16);
                sacc[t] = __builtin_amdgcn_wmma_f32_16x16x32_bf16(false, qcf[c], false, kf,
                                                                  (short)0, sacc[t], false, false);
                v16bf pf2 = *(const v16bf*)(prow + c * 32 + hi * 16);
                sacc[t] = __builtin_amdgcn_wmma_f32_16x16x32_bf16(false, qpf[c], false, pf2,
                                                                  (short)0, sacc[t], false, false);
            }
        }

        // --- Scale + causal mask + online softmax update ---
        float pv[2][8];
        #pragma unroll
        for (int r = 0; r < 8; ++r) {
            const int j = jt * 16 + r + hi * 8;
            float s0 = sacc[0][r] * 0.125f;
            float s1 = sacc[1][r] * 0.125f;
            if (kb + lo > j)      s0 = -1e30f;
            if (kb + 16 + lo > j) s1 = -1e30f;
            pv[0][r] = s0; pv[1][r] = s1;
        }
        #pragma unroll
        for (int r = 0; r < 8; ++r) {
            float bm = fmaxf(pv[0][r], pv[1][r]);
            bm = fmaxf(bm, __shfl_xor(bm, 1, 32));
            bm = fmaxf(bm, __shfl_xor(bm, 2, 32));
            bm = fmaxf(bm, __shfl_xor(bm, 4, 32));
            bm = fmaxf(bm, __shfl_xor(bm, 8, 32));
            const float mnew  = fmaxf(mrun[r], bm);
            const float scale = __expf(mrun[r] - mnew);
            const float p0 = __expf(pv[0][r] - mnew);
            const float p1 = __expf(pv[1][r] - mnew);
            float bs = p0 + p1;
            bs += __shfl_xor(bs, 1, 32);
            bs += __shfl_xor(bs, 2, 32);
            bs += __shfl_xor(bs, 4, 32);
            bs += __shfl_xor(bs, 8, 32);
            lrun[r] = lrun[r] * scale + bs;
            mrun[r] = mnew;
            pv[0][r] = p0; pv[1][r] = p1;
            #pragma unroll
            for (int g = 0; g < 4; ++g) oacc[g][r] *= scale;
        }

        // --- P through LDS (D-layout -> A-layout) ---
        __syncthreads();
        #pragma unroll
        for (int t = 0; t < 2; ++t)
            #pragma unroll
            for (int r = 0; r < 8; ++r)
                p_lds[(r + hi * 8) * 32 + t * 16 + lo] = f2bf(pv[t][r]);
        __syncthreads();

        v16bf pf;
        {
            v8bf l0 = *(const v8bf*)&p_lds[lo * 32 + hi * 16];
            v8bf l1 = *(const v8bf*)&p_lds[lo * 32 + hi * 16 + 8];
            #pragma unroll
            for (int i = 0; i < 8; ++i) { pf[i] = l0[i]; pf[8 + i] = l1[i]; }
        }
        // --- O += P @ V : B fragment is a contiguous slice of d-major V ---
        #pragma unroll
        for (int g = 0; g < 4; ++g) {
            v16bf vf = *(const v16bf*)(vt + headBase + (size_t)(g * 16 + lo) * S_LEN + kb + hi * 16);
            oacc[g] = __builtin_amdgcn_wmma_f32_16x16x32_bf16(false, pf, false, vf,
                                                              (short)0, oacc[g], false, false);
        }
    }

    // --- Normalize and store (B, S, EMB) f32 ---
    float* obase = outp + (size_t)b * S_LEN * EMBD;
    #pragma unroll
    for (int g = 0; g < 4; ++g)
        #pragma unroll
        for (int r = 0; r < 8; ++r) {
            const int j = jt * 16 + r + hi * 8;
            const int e = h * HDIM + g * 16 + lo;
            obase[(size_t)j * EMBD + e] = oacc[g][r] / lrun[r];
        }
}

// ---------------------------------------------------------------------------
// Residual + LayerNorm (no affine), eps 1e-5.  grid.y selects u/f stream.
// ---------------------------------------------------------------------------
__global__ __launch_bounds__(128)
void ln_kernel(const float* __restrict__ u_emb, const float* __restrict__ att_u,
               const float* __restrict__ f_emb, const float* __restrict__ att_f,
               float* __restrict__ out)
{
    __shared__ float red[128];
    const int row = blockIdx.x;
    const int st  = blockIdx.y;
    const float* emb = st ? f_emb : u_emb;
    const float* att = st ? att_f : att_u;
    const size_t base = (size_t)row * EMBD;
    const int tid = threadIdx.x;

    float x[4];
    float s = 0.0f;
    #pragma unroll
    for (int i = 0; i < 4; ++i) {
        const int idx = tid + i * 128;
        x[i] = emb[base + idx] + att[base + idx];
        s += x[i];
    }
    red[tid] = s;
    __syncthreads();
    for (int off = 64; off > 0; off >>= 1) {
        if (tid < off) red[tid] += red[tid + off];
        __syncthreads();
    }
    const float mean = red[0] / (float)EMBD;
    __syncthreads();

    float q = 0.0f;
    #pragma unroll
    for (int i = 0; i < 4; ++i) { const float d = x[i] - mean; q += d * d; }
    red[tid] = q;
    __syncthreads();
    for (int off = 64; off > 0; off >>= 1) {
        if (tid < off) red[tid] += red[tid + off];
        __syncthreads();
    }
    const float var  = red[0] / (float)EMBD;
    const float rstd = rsqrtf(var + 1e-5f);

    float* o = out + (size_t)st * ((size_t)MROWS * EMBD) + base;
    #pragma unroll
    for (int i = 0; i < 4; ++i)
        o[tid + i * 128] = (x[i] - mean) * rstd;
}

// ---------------------------------------------------------------------------
extern "C" void kernel_launch(void* const* d_in, const int* in_sizes, int n_in,
                              void* d_out, int out_size, void* d_ws, size_t ws_size,
                              hipStream_t stream)
{
    (void)in_sizes; (void)n_in; (void)out_size; (void)ws_size;

    const float* u_emb = (const float*)d_in[0];
    const float* f_emb = (const float*)d_in[1];
    const float* gpe   = (const float*)d_in[2];
    // d_in[3], d_in[4]: causal masks, handled analytically
    const float* Wq_w  = (const float*)d_in[5];
    const float* Wq_b  = (const float*)d_in[6];
    const float* Wkv_w = (const float*)d_in[7];
    const float* Wkv_b = (const float*)d_in[8];
    const float* Wp_w  = (const float*)d_in[9];
    const float* Wp_b  = (const float*)d_in[10];
    const float* Wu_w  = (const float*)d_in[11];
    const float* Wu_b  = (const float*)d_in[12];
    const float* Bfc   = (const float*)d_in[13];
    const float* Bfp   = (const float*)d_in[14];
    const float* Buc   = (const float*)d_in[15];
    const float* Bup   = (const float*)d_in[16];

    char* ws = (char*)d_ws;
    const size_t MB = 1024 * 1024;
    __bf16* qc_u  = (__bf16*)(ws +  0 * MB);   // (B*NH, S, 64)  2 MB
    __bf16* qp_u  = (__bf16*)(ws +  2 * MB);
    __bf16* qc_f  = (__bf16*)(ws +  4 * MB);
    __bf16* qp_f  = (__bf16*)(ws +  6 * MB);
    __bf16* kh    = (__bf16*)(ws +  8 * MB);
    __bf16* vt    = (__bf16*)(ws + 10 * MB);   // (B*NH, 64, S)  d-major
    __bf16* kp    = (__bf16*)(ws + 12 * MB);
    float*  att_u = (float*) (ws + 14 * MB);   // (B, S, 512)    4 MB
    float*  att_f = (float*) (ws + 18 * MB);
    __bf16* a_bf  = (__bf16*)(ws + 22 * MB);   // f_emb bf16     2 MB
    __bf16* g_bf  = (__bf16*)(ws + 24 * MB);   // gpe bf16       0.5 MB
    __bf16* wt_u  = (__bf16*)(ws + 25 * MB);   // 512x512        0.5 MB
    __bf16* wt_q  = (__bf16*)(ws + 26 * MB);
    __bf16* wt_kv = (__bf16*)(ws + 27 * MB);   // 1024x512       1 MB
    __bf16* wt_p  = (__bf16*)(ws + 28 * MB);   // 512x128        .125 MB

    // --- Prep: bf16 conversion + weight transposes (one-time traffic) ---
    cvt_bf16_kernel<<<dim3((MROWS * EMBD) / 4 / 256), dim3(256), 0, stream>>>(f_emb, a_bf);
    cvt_bf16_kernel<<<dim3((MROWS * 128) / 4 / 256), dim3(256), 0, stream>>>(gpe, g_bf);
    transpose_bf16_kernel<<<dim3(16, 16), dim3(256), 0, stream>>>(Wu_w,  wt_u,  EMBD, EMBD);
    transpose_bf16_kernel<<<dim3(16, 16), dim3(256), 0, stream>>>(Wq_w,  wt_q,  EMBD, EMBD);
    transpose_bf16_kernel<<<dim3(32, 16), dim3(256), 0, stream>>>(Wkv_w, wt_kv, EMBD, 2 * EMBD);
    transpose_bf16_kernel<<<dim3(16,  4), dim3(256), 0, stream>>>(Wp_w,  wt_p,  128,  EMBD);

    const dim3 wv(32);
    // q_u = f_emb@Wu (+Buc/+Bup, tanh)  -> qc_u, qp_u
    proj_gemm_kernel<<<dim3(EMBD / 64, MROWS / 16), wv, 0, stream>>>(
        a_bf, wt_u, Wu_b, Buc, Bup, qc_u, qp_u, EMBD, 0);
    // q_f = f_emb@Wq (+Bfc/+Bfp, tanh)  -> qc_f, qp_f
    proj_gemm_kernel<<<dim3(EMBD / 64, MROWS / 16), wv, 0, stream>>>(
        a_bf, wt_q, Wq_b, Bfc, Bfp, qc_f, qp_f, EMBD, 1);
    // kv = f_emb@Wkv, split + tanh      -> kh (row-major), vt (d-major)
    proj_gemm_kernel<<<dim3(2 * EMBD / 64, MROWS / 16), wv, 0, stream>>>(
        a_bf, wt_kv, Wkv_b, nullptr, nullptr, kh, vt, EMBD, 2);
    // kp = tanh(gpe@Wp + b)             -> kp
    proj_gemm_kernel<<<dim3(EMBD / 64, MROWS / 16), wv, 0, stream>>>(
        g_bf, wt_p, Wp_b, nullptr, nullptr, kp, nullptr, 128, 3);

    attn_kernel<<<dim3(S_LEN / 16, NHEAD, BATCH), wv, 0, stream>>>(
        qc_u, qp_u, kh, kp, vt, att_u);
    attn_kernel<<<dim3(S_LEN / 16, NHEAD, BATCH), wv, 0, stream>>>(
        qc_f, qp_f, kh, kp, vt, att_f);

    ln_kernel<<<dim3(MROWS, 2), dim3(128), 0, stream>>>(
        u_emb, att_u, f_emb, att_f, (float*)d_out);
}
